// AttentionalSpikingSSMLayer_60000693125490
// MI455X (gfx1250) — compile-verified, matrix operand order checked by reading, not run on hardware
//
#include <hip/hip_runtime.h>
#include <hip/hip_bf16.h>

// CDNA5 / gfx1250, wave32: bf16 WMMA + async global->LDS (ASYNCcnt) pipeline.

#define BB   8
#define TT   16
#define SS   256
#define DD   512
#define DSD  64
#define HH   4
#define DH   16
#define NROW 2048           // B*S
#define MEM_DECAY 0.6065306597126334f   // exp(-1/2)
#define ADAPT 0.1f
#define TARGET 0.02f
#define THRMIN 0.5f

typedef __attribute__((ext_vector_type(16))) __bf16 v16bf;
typedef __attribute__((ext_vector_type(8)))  float  v8f;
typedef unsigned int u32;
typedef unsigned short u16;

union FragU {            // one WMMA 16-element bf16 operand = 32 bytes
  uint4 q[2];
  u32   w[8];
  v16bf v;
};

static __device__ inline u32 pack_bf16(float a, float b) {
  union { __bf16 h[2]; u32 u; } c;
  c.h[0] = (__bf16)a; c.h[1] = (__bf16)b;
  return c.u;            // lowers to v_cvt_pk_bf16_f32
}
static __device__ inline u16 bf16_bits(float a) {
  union { __bf16 h; u16 u; } c;
  c.h = (__bf16)a;
  return c.u;
}

// ---------------------------------------------------------------------------
// Generic tiled GEMM:  out[M=2048, Nout] = X[M,K] @ W[Nout,K]^T (+bias)(+addin)
// XMODE==0: X row-major [row*K + k]
// XMODE==1: X is x[:, t] slice of (B,T,S,D): row=(b*S+s) -> ((b*T+t)*S+s)*D + k
//
// 8 waves/block, one 16x16 tile per wave. K chunks of 32 are DMA'd into
// double-buffered *wave-private* LDS tiles with global_load_async_to_lds_b128
// (ASYNCcnt), overlapping the next chunk's copy with the current WMMA.
// No __syncthreads needed: each wave owns its slab. bf16 conversion happens
// during the fragment build (ds_load_b128 + v_cvt_pk_bf16_f32).
// ---------------------------------------------------------------------------
template<int XMODE, bool HASB, bool HASADD>
__global__ __launch_bounds__(256)
void gemm_wmma_kernel(const float* __restrict__ X, int t, int K,
                      const float* __restrict__ W, const float* __restrict__ bias,
                      const float* __restrict__ addin, float* __restrict__ out,
                      int Nout, int tilesN) {
  __shared__ __align__(16) float Af[8][2][16][32];  // [wave][buf][row][k]
  __shared__ __align__(16) float Bf[8][2][16][32];  // [wave][buf][n][k] (W rows)
  const int wave = threadIdx.x >> 5;
  const int lane = threadIdx.x & 31;
  const int tile = blockIdx.x * 8 + wave;
  const int tm = tile / tilesN;
  const int tn = tile % tilesN;
  const int nloc = lane & 15;
  const bool lo = lane < 16;

  // LDS byte offsets for the async DMA destinations (flat addr low 32 bits)
  u32 abase[2], bbase[2];
  abase[0] = (u32)(uintptr_t)&Af[wave][0][0][0];
  abase[1] = (u32)(uintptr_t)&Af[wave][1][0][0];
  bbase[0] = (u32)(uintptr_t)&Bf[wave][0][0][0];
  bbase[1] = (u32)(uintptr_t)&Bf[wave][1][0][0];

  // segment mapping: tile = 16 rows x 32 f32 = 128 x 16B segments;
  // per async instruction each lane moves one b128 segment.
  const int seg0 = lane;               // this lane's first segment
  auto xrowoff = [&](int row) -> long {
    if (XMODE == 0) return (long)row * K;
    int b = row >> 8, s = row & 255;
    return ((long)(b * TT + t) * SS + s) * DD;
  };

  v8f acc = {};
  int buf = 0;

  // issue async copies for chunk at k0 into buffer b (8 instructions/wave)
#define ISSUE_CHUNK(k0_, b_)                                                   \
  {                                                                            \
    _Pragma("unroll")                                                          \
    for (int i = 0; i < 4; ++i) {                                              \
      int seg = i * 32 + seg0;                                                 \
      int row = seg >> 3, sj = seg & 7;                                        \
      const float* ga = X + xrowoff(tm * 16 + row) + (k0_) + sj * 4;           \
      u32 la = abase[b_] + (u32)seg * 16;                                      \
      asm volatile("global_load_async_to_lds_b128 %0, %1, off"                 \
                   :: "v"(la), "v"(ga) : "memory");                            \
      const float* gb = W + (long)(tn * 16 + row) * K + (k0_) + sj * 4;        \
      u32 lb = bbase[b_] + (u32)seg * 16;                                      \
      asm volatile("global_load_async_to_lds_b128 %0, %1, off"                 \
                   :: "v"(lb), "v"(gb) : "memory");                            \
    }                                                                          \
  }

  ISSUE_CHUNK(0, 0);
  for (int k0 = 0; k0 < K; k0 += 32) {
    const bool more = (k0 + 32) < K;
    if (more) {
      ISSUE_CHUNK(k0 + 32, buf ^ 1);
      asm volatile("s_wait_asynccnt 0x8" ::: "memory");  // current buf resident
    } else {
      asm volatile("s_wait_asynccnt 0x0" ::: "memory");
    }
    // fragment build: lane m/n = lane&15; lanes<16 K=0..15, lanes>=16 K=16..31
    const int kb = lo ? 0 : 16;
    FragU a, b;
    const float4* ar = (const float4*)&Af[wave][buf][nloc][kb];
    const float4* br = (const float4*)&Bf[wave][buf][nloc][kb];
#pragma unroll
    for (int i = 0; i < 4; ++i) {
      float4 fa = ar[i];
      a.w[2 * i]     = pack_bf16(fa.x, fa.y);
      a.w[2 * i + 1] = pack_bf16(fa.z, fa.w);
      float4 fb = br[i];
      b.w[2 * i]     = pack_bf16(fb.x, fb.y);
      b.w[2 * i + 1] = pack_bf16(fb.z, fb.w);
    }
    acc = __builtin_amdgcn_wmma_f32_16x16x32_bf16(false, a.v, false, b.v,
                                                  (short)0, acc, false, false);
    buf ^= 1;
  }
#undef ISSUE_CHUNK

  // D layout: VGPR r holds M = r + (lane<16 ? 0 : 8), N = lane&15
#pragma unroll
  for (int r = 0; r < 8; ++r) {
    int row = tm * 16 + r + (lo ? 0 : 8);
    int col = tn * 16 + nloc;
    float v = acc[r];
    if (HASB)   v += bias[col];
    if (HASADD) v += addin[(long)row * Nout + col];
    out[(long)row * Nout + col] = v;
  }
}

// ---------------------------------------------------------------------------
// Flash attention: H=4 heads, N=2048, dh=16 (K padded to 32 for WMMA).
// One wave per (head, 16-query block); per key block j: S-tile WMMA, online
// softmax via half-wave shuffles, P-tile -> bf16 LDS -> A-layout reload,
// V-tile staged transposed-bf16 in LDS, P@V WMMA accumulate.
// qb: (2048,64) pre-projected q.  kvb: (2048,128) = [k | v].
// ---------------------------------------------------------------------------
__global__ __launch_bounds__(256)
void flash_attn_kernel(const float* __restrict__ qb, const float* __restrict__ kvb,
                       float* __restrict__ att) {
  __shared__ __align__(16) u16 Plds[8][16][16];  // [wave][m][n] bf16
  __shared__ __align__(16) u32 VsT[8][16][8];    // [wave][n][kpair] bf16 x2
  const int wave = threadIdx.x >> 5;
  const int lane = threadIdx.x & 31;
  const int wid  = blockIdx.x * 8 + wave;
  const int head = wid >> 7;   // 0..3
  const int qblk = wid & 127;  // 0..127
  const int nloc = lane & 15;
  const bool lo  = lane < 16;
  const float scale = 0.25f;   // 1/sqrt(dh=16)

  // Q fragment (A layout, scaled); lanes>=16 carry K=16..31 zero padding
  FragU qa;
  {
    const float4* qrow = (const float4*)(qb + (long)(qblk * 16 + nloc) * DSD + head * DH);
    if (lo) {
      float4 f0 = qrow[0], f1 = qrow[1], f2 = qrow[2], f3 = qrow[3];
      qa.q[0] = uint4{pack_bf16(f0.x * scale, f0.y * scale),
                      pack_bf16(f0.z * scale, f0.w * scale),
                      pack_bf16(f1.x * scale, f1.y * scale),
                      pack_bf16(f1.z * scale, f1.w * scale)};
      qa.q[1] = uint4{pack_bf16(f2.x * scale, f2.y * scale),
                      pack_bf16(f2.z * scale, f2.w * scale),
                      pack_bf16(f3.x * scale, f3.y * scale),
                      pack_bf16(f3.z * scale, f3.w * scale)};
    } else {
      qa.q[0] = uint4{0, 0, 0, 0};
      qa.q[1] = uint4{0, 0, 0, 0};
    }
  }

  v8f o = {};
  float mrow[8], lrow[8];
#pragma unroll
  for (int r = 0; r < 8; ++r) { mrow[r] = -1e30f; lrow[r] = 0.0f; }

  for (int j = 0; j < NROW / 16; ++j) {
    // K fragment (B layout): lane n = key j*16+n, 16 contiguous f32 -> bf16
    FragU kb;
    if (lo) {
      const float4* krow = (const float4*)(kvb + (long)(j * 16 + nloc) * 128 + head * DH);
      float4 f0 = krow[0], f1 = krow[1], f2 = krow[2], f3 = krow[3];
      kb.q[0] = uint4{pack_bf16(f0.x, f0.y), pack_bf16(f0.z, f0.w),
                      pack_bf16(f1.x, f1.y), pack_bf16(f1.z, f1.w)};
      kb.q[1] = uint4{pack_bf16(f2.x, f2.y), pack_bf16(f2.z, f2.w),
                      pack_bf16(f3.x, f3.y), pack_bf16(f3.z, f3.w)};
    } else {
      kb.q[0] = uint4{0, 0, 0, 0};
      kb.q[1] = uint4{0, 0, 0, 0};
    }

    v8f s = {};
    s = __builtin_amdgcn_wmma_f32_16x16x32_bf16(false, qa.v, false, kb.v,
                                                (short)0, s, false, false);

    // stage V tile transposed bf16: VsT[n][k] ; 4 pair-loads per lane
#pragma unroll
    for (int i = 0; i < 4; ++i) {
      int p = lane + i * 32;          // 0..127 pair index
      int n = p >> 3, kp = p & 7;
      float va = kvb[(long)(j * 16 + 2 * kp    ) * 128 + DSD + head * DH + n];
      float vb_ = kvb[(long)(j * 16 + 2 * kp + 1) * 128 + DSD + head * DH + n];
      VsT[wave][n][kp] = pack_bf16(va, vb_);
    }
    if (j + 1 < NROW / 16)  // prefetch next key block rows (global_prefetch_b8)
      __builtin_prefetch(kvb + (long)((j + 1) * 16 + nloc) * 128, 0, 3);

    // online softmax per row (rows split across lane halves; masks<=8 in-half)
#pragma unroll
    for (int r = 0; r < 8; ++r) {
      float sv = s[r];
      float rmax = sv;
#pragma unroll
      for (int m = 1; m < 16; m <<= 1) rmax = fmaxf(rmax, __shfl_xor(rmax, m, 32));
      float mnew = fmaxf(mrow[r], rmax);
      float p = __expf(sv - mnew);
      float psum = p;
#pragma unroll
      for (int m = 1; m < 16; m <<= 1) psum += __shfl_xor(psum, m, 32);
      float alpha = __expf(mrow[r] - mnew);
      lrow[r] = lrow[r] * alpha + psum;
      mrow[r] = mnew;
      o[r] *= alpha;
      Plds[wave][r + (lo ? 0 : 8)][nloc] = bf16_bits(p);  // D layout -> LDS
    }
    __syncthreads();
    // P fragment (A layout) and V fragment (B layout): contiguous b128 loads
    FragU pa, vb;
    if (lo) {
      pa.q[0] = *(const uint4*)&Plds[wave][nloc][0];
      pa.q[1] = *(const uint4*)&Plds[wave][nloc][8];
      vb.q[0] = *(const uint4*)&VsT[wave][nloc][0];
      vb.q[1] = *(const uint4*)&VsT[wave][nloc][4];
    } else {
      pa.q[0] = uint4{0, 0, 0, 0}; pa.q[1] = uint4{0, 0, 0, 0};
      vb.q[0] = uint4{0, 0, 0, 0}; vb.q[1] = uint4{0, 0, 0, 0};
    }
    o = __builtin_amdgcn_wmma_f32_16x16x32_bf16(false, pa.v, false, vb.v,
                                                (short)0, o, false, false);
    __syncthreads();
  }
#pragma unroll
  for (int r = 0; r < 8; ++r) {
    int row = qblk * 16 + r + (lo ? 0 : 8);
    att[(long)row * DSD + head * DH + nloc] = o[r] / lrow[r];
  }
}

// ---------------------------------------------------------------------------
// LIF step on state (2048x64): spike/heaviside, membrane reset, spike count.
// ---------------------------------------------------------------------------
__global__ __launch_bounds__(256)
void lif_state_kernel(const float* __restrict__ u, float* __restrict__ sv,
                      float* __restrict__ h, const float* __restrict__ thrp,
                      float* __restrict__ sump, int n) {
  int idx = blockIdx.x * 256 + threadIdx.x;
  float thr = thrp[0];
  float spike = 0.0f;
  if (idx < n) {
    float vpot = sv[idx] * MEM_DECAY + u[idx];
    spike = (vpot - thr >= 0.0f) ? 1.0f : 0.0f;
    sv[idx] = vpot * (1.0f - spike);
    h[idx] = spike;
  }
  float s = spike;
#pragma unroll
  for (int m = 1; m < 32; m <<= 1) s += __shfl_xor(s, m, 32);
  if ((threadIdx.x & 31) == 0) atomicAdd(sump, s);
}

// LIF on output potentials (2048x512) -> writes spikes to out[:, t] slice.
__global__ __launch_bounds__(256)
void lif_out_kernel(const float* __restrict__ pot, float* __restrict__ ov,
                    const float* __restrict__ thrp, float* __restrict__ sump,
                    float* __restrict__ out, int t) {
  int idx = blockIdx.x * 256 + threadIdx.x;  // grid exactly covers 2048*512
  float thr = thrp[0];
  float vpot = ov[idx] * MEM_DECAY + pot[idx];
  float spike = (vpot - thr >= 0.0f) ? 1.0f : 0.0f;
  ov[idx] = vpot * (1.0f - spike);
  int row = idx >> 9, d = idx & 511;
  int b = row >> 8, s = row & 255;
  out[(((long)(b * TT + t) * SS + s) * DD) + d] = spike;
  float sm = spike;
#pragma unroll
  for (int m = 1; m < 32; m <<= 1) sm += __shfl_xor(sm, m, 32);
  if ((threadIdx.x & 31) == 0) atomicAdd(sump, sm);
}

// Threshold adaptation (thresholds stay uniform -> scalar), reset counter.
__global__ void thr_update_kernel(float* thr, float* sum, float invn) {
  if (threadIdx.x == 0 && blockIdx.x == 0) {
    float err = sum[0] * invn - TARGET;
    thr[0] = fmaxf(thr[0] + ADAPT * err, THRMIN);
    sum[0] = 0.0f;
  }
}

// Zero carried state (h, sv, ov contiguous at ws start) + init scalars.
__global__ __launch_bounds__(256)
void init_kernel(float* ws, const float* thr_s0, const float* thr_o0,
                 int nzero, long scal_off) {
  int idx = blockIdx.x * 256 + threadIdx.x;
  if (idx < nzero) ws[idx] = 0.0f;
  if (idx == 0) {
    ws[scal_off + 0] = thr_s0[0];  // thr_s
    ws[scal_off + 1] = thr_o0[0];  // thr_o
    ws[scal_off + 2] = 0.0f;       // sum_s
    ws[scal_off + 3] = 0.0f;       // sum_o
  }
}

// ---------------------------------------------------------------------------
// Workspace layout (floats): h@0, sv@131072, ov@262144, kv@1310720,
// qb@1572864, st@1703936, att@1835008, u@1966080, pot@2097152,
// scalars@3145728 (thr_s, thr_o, sum_s, sum_o). Total ~12.6 MB.
// ---------------------------------------------------------------------------
extern "C" void kernel_launch(void* const* d_in, const int* in_sizes, int n_in,
                              void* d_out, int out_size, void* d_ws, size_t ws_size,
                              hipStream_t stream) {
  const float* x      = (const float*)d_in[0];
  const float* A      = (const float*)d_in[1];
  const float* C      = (const float*)d_in[2];
  const float* Wq     = (const float*)d_in[3];
  const float* bq     = (const float*)d_in[4];
  const float* Wkv    = (const float*)d_in[5];
  const float* bkv    = (const float*)d_in[6];
  const float* Wo     = (const float*)d_in[7];
  const float* bo     = (const float*)d_in[8];
  const float* thr_s0 = (const float*)d_in[9];
  const float* thr_o0 = (const float*)d_in[10];
  float* out = (float*)d_out;
  float* ws  = (float*)d_ws;

  float* h    = ws + 0;
  float* sv   = ws + 131072;
  float* ov   = ws + 262144;
  float* kv   = ws + 1310720;
  float* qb   = ws + 1572864;
  float* st   = ws + 1703936;
  float* att  = ws + 1835008;
  float* u    = ws + 1966080;
  float* pot  = ws + 2097152;
  const long scal = 3145728;
  float* thr_s = ws + scal + 0;
  float* thr_o = ws + scal + 1;
  float* sum_s = ws + scal + 2;
  float* sum_o = ws + scal + 3;

  const int nzero = 131072 + 131072 + 1048576;  // h, sv, ov
  init_kernel<<<(nzero + 255) / 256, 256, 0, stream>>>(ws, thr_s0, thr_o0, nzero, scal);

  for (int t = 0; t < TT; ++t) {
    // kv = x_t @ Wkv^T + bkv   (2048 x 128, K=512)
    gemm_wmma_kernel<1, true, false><<<128, 256, 0, stream>>>(
        x, t, DD, Wkv, bkv, nullptr, kv, 128, 8);
    // st = h @ A^T             (2048 x 64, K=64)
    gemm_wmma_kernel<0, false, false><<<64, 256, 0, stream>>>(
        h, 0, DSD, A, nullptr, nullptr, st, DSD, 4);
    // q = h @ Wq^T + bq
    gemm_wmma_kernel<0, true, false><<<64, 256, 0, stream>>>(
        h, 0, DSD, Wq, bq, nullptr, qb, DSD, 4);
    // att = softmax(q k^T / sqrt(dh)) v   (flash, per-head)
    flash_attn_kernel<<<64, 256, 0, stream>>>(qb, kv, att);
    // u = st + att @ Wo^T + bo
    gemm_wmma_kernel<0, true, true><<<64, 256, 0, stream>>>(
        att, 0, DSD, Wo, bo, st, u, DSD, 4);
    // LIF #1 -> h (spikes), sv, spike count; threshold update
    lif_state_kernel<<<512, 256, 0, stream>>>(u, sv, h, thr_s, sum_s, NROW * DSD);
    thr_update_kernel<<<1, 1, 0, stream>>>(thr_s, sum_s, 1.0f / (NROW * DSD));
    // pot = h @ C^T            (2048 x 512, K=64)
    gemm_wmma_kernel<0, false, false><<<512, 256, 0, stream>>>(
        h, 0, DSD, C, nullptr, nullptr, pot, DD, 32);
    // LIF #2 -> out[:, t], ov, count; threshold update
    lif_out_kernel<<<(NROW * DD) / 256, 256, 0, stream>>>(pot, ov, thr_o, sum_o,
                                                          out, t);
    thr_update_kernel<<<1, 1, 0, stream>>>(thr_o, sum_o, 1.0f / (NROW * DD));
  }
}
